// QuadraticFormSheafLearner_81484119540403
// MI455X (gfx1250) — matrix operand order repaired and checked by reading, not compiled, not run on hardware
//
#include <hip/hip_runtime.h>
#include <hip/hip_bf16.h>
#include <math.h>

typedef __attribute__((ext_vector_type(16))) __bf16 v16bf;
typedef __attribute__((ext_vector_type(8)))  float  v8f;

#define C_DIM 64
#define D_DIM 16

// ---------------------------------------------------------------------------
// Phase 1: P = x @ W[:64], Q = x @ W[64:]   via bf16-split WMMA (f32 accum).
// One wave per 16-node tile. 12x v_wmma_f32_16x16x32_bf16 per tile.
// ---------------------------------------------------------------------------
__global__ __launch_bounds__(256) void pq_wmma_kernel(
    const float* __restrict__ x, const float* __restrict__ W,
    float* __restrict__ P, float* __restrict__ Q, int N) {
  const int lane = threadIdx.x & 31;
  const int wave = threadIdx.x >> 5;
  const int tile = blockIdx.x * 8 + wave;
  const int base = tile * 16;
  if (base >= N) return;            // uniform per wave: EXEC stays all-ones

  const int m    = lane & 15;
  const int half = lane >> 4;       // 0: lanes 0-15, 1: lanes 16-31
  int node = base + m;
  if (node >= N) node = N - 1;      // N % 16 == 0 in practice; keep EXEC full
  const float* __restrict__ xrow = x + (size_t)node * C_DIM;

  // ---- A fragments (16x32 bf16 per K-chunk), split into hi/lo bf16 ----
  // Contiguous K order, same as B: lane<16 -> K = 0..15; lane>=16 -> K = 16..31
  v16bf ah[2], al[2];
#pragma unroll
  for (int c = 0; c < 2; ++c) {
#pragma unroll
    for (int i = 0; i < 16; ++i) {
      const int k = half * 16 + i + 32 * c;
      const float v  = xrow[k];
      const __bf16 h = (__bf16)v;
      ah[c][i] = h;
      al[c][i] = (__bf16)(v - (float)h);
    }
  }

  // ---- B fragments (32x16 bf16 per K-chunk) for W_top and W_bot ----
  // Contiguous K order, same as A: lane<16 -> K = 0..15; lane>=16 -> K = 16..31, column = lane&15
  const int col = lane & 15;
  const int kb  = half * 16;
  v16bf bth[2], btl[2], bbh[2], bbl[2];
#pragma unroll
  for (int c = 0; c < 2; ++c) {
#pragma unroll
    for (int i = 0; i < 16; ++i) {
      const int k = kb + i + 32 * c;
      const float wt = W[(size_t)k * D_DIM + col];            // W_top[k][col]
      const float wb = W[(size_t)(C_DIM + k) * D_DIM + col];  // W_bot[k][col]
      const __bf16 th = (__bf16)wt;
      const __bf16 bh = (__bf16)wb;
      bth[c][i] = th;  btl[c][i] = (__bf16)(wt - (float)th);
      bbh[c][i] = bh;  bbl[c][i] = (__bf16)(wb - (float)bh);
    }
  }

  // ---- bf16x3 accumulation: hi*hi + hi*lo + lo*hi in f32 ----
  v8f accP = {};
  v8f accQ = {};
#pragma unroll
  for (int c = 0; c < 2; ++c) {
    accP = __builtin_amdgcn_wmma_f32_16x16x32_bf16(false, ah[c], false, bth[c],
                                                   (short)0, accP, false, false);
    accP = __builtin_amdgcn_wmma_f32_16x16x32_bf16(false, ah[c], false, btl[c],
                                                   (short)0, accP, false, false);
    accP = __builtin_amdgcn_wmma_f32_16x16x32_bf16(false, al[c], false, bth[c],
                                                   (short)0, accP, false, false);
  }
#pragma unroll
  for (int c = 0; c < 2; ++c) {
    accQ = __builtin_amdgcn_wmma_f32_16x16x32_bf16(false, ah[c], false, bbh[c],
                                                   (short)0, accQ, false, false);
    accQ = __builtin_amdgcn_wmma_f32_16x16x32_bf16(false, ah[c], false, bbl[c],
                                                   (short)0, accQ, false, false);
    accQ = __builtin_amdgcn_wmma_f32_16x16x32_bf16(false, al[c], false, bbh[c],
                                                   (short)0, accQ, false, false);
  }

  // ---- store D (16x16 f32): VGPR r -> row r (lanes 0-15) / row r+8 ----
#pragma unroll
  for (int r = 0; r < 8; ++r) {
    const int row = base + r + half * 8;
    if (row < N) {
      P[(size_t)row * D_DIM + col] = accP[r];
      Q[(size_t)row * D_DIM + col] = accQ[r];
    }
  }
}

// ---------------------------------------------------------------------------
// Phase 2: out[e] = tanh(P[row[e]] + Q[col[e]]).  4 threads per edge, float4.
// P/Q (6.4 MB each) are L2-resident -> gathers hit L2, writes stream to HBM.
// ---------------------------------------------------------------------------
__global__ __launch_bounds__(256) void edge_tanh_kernel(
    const int* __restrict__ ei, const float* __restrict__ P,
    const float* __restrict__ Q, float* __restrict__ out, int E) {
  const long long t = (long long)blockIdx.x * blockDim.x + threadIdx.x;
  const long long e = t >> 2;
  const int sub = (int)(t & 3);
  if (e >= E) return;
  const int r = ei[e];
  const int c = ei[(size_t)E + e];
  const float4 p = reinterpret_cast<const float4*>(P + (size_t)r * D_DIM)[sub];
  const float4 q = reinterpret_cast<const float4*>(Q + (size_t)c * D_DIM)[sub];
  float4 o;
  o.x = tanhf(p.x + q.x);
  o.y = tanhf(p.y + q.y);
  o.z = tanhf(p.z + q.z);
  o.w = tanhf(p.w + q.w);
  reinterpret_cast<float4*>(out + (size_t)e * D_DIM)[sub] = o;
}

// ---------------------------------------------------------------------------
// Fallback (workspace too small): direct per-(edge, output) dot products.
// ---------------------------------------------------------------------------
__global__ __launch_bounds__(256) void fused_fallback_kernel(
    const float* __restrict__ x, const int* __restrict__ ei,
    const float* __restrict__ W, float* __restrict__ out, int E) {
  const long long t = (long long)blockIdx.x * blockDim.x + threadIdx.x;
  const long long e = t >> 4;
  const int j = (int)(t & 15);
  if (e >= E) return;
  const int r = ei[e];
  const int c = ei[(size_t)E + e];
  const float* __restrict__ xr = x + (size_t)r * C_DIM;
  const float* __restrict__ xc = x + (size_t)c * C_DIM;
  float s = 0.0f;
#pragma unroll 8
  for (int k = 0; k < C_DIM; ++k) {
    s = fmaf(xr[k], W[(size_t)k * D_DIM + j], s);
    s = fmaf(xc[k], W[(size_t)(C_DIM + k) * D_DIM + j], s);
  }
  out[(size_t)e * D_DIM + j] = tanhf(s);
}

extern "C" void kernel_launch(void* const* d_in, const int* in_sizes, int n_in,
                              void* d_out, int out_size, void* d_ws, size_t ws_size,
                              hipStream_t stream) {
  const float* x  = (const float*)d_in[0];
  const int*   ei = (const int*)d_in[1];
  const float* W  = (const float*)d_in[2];
  float* out = (float*)d_out;

  const int N = in_sizes[0] / C_DIM;   // 100000
  const int E = in_sizes[1] / 2;       // 1600000

  const size_t pq_bytes = (size_t)N * D_DIM * sizeof(float);
  if (ws_size >= 2 * pq_bytes) {
    float* P = (float*)d_ws;
    float* Q = P + (size_t)N * D_DIM;

    const int tiles  = (N + 15) / 16;
    const int blocks = (tiles + 7) / 8;    // 8 waves (tiles) per 256-thread block
    pq_wmma_kernel<<<blocks, 256, 0, stream>>>(x, W, P, Q, N);

    const long long threads = (long long)E * 4;
    const int eb = (int)((threads + 255) / 256);
    edge_tanh_kernel<<<eb, 256, 0, stream>>>(ei, P, Q, out, E);
  } else {
    const long long threads = (long long)E * 16;
    const int fb = (int)((threads + 255) / 256);
    fused_fallback_kernel<<<fb, 256, 0, stream>>>(x, ei, W, out, E);
  }
}
